// ScaledDotProductAttention_19481971655138
// MI455X (gfx1250) — matrix operand order repaired and checked by reading, not compiled, not run on hardware
//
#include <hip/hip_runtime.h>

// Flash attention (causal) for B=2, H=16, S=2048, D=128, fp32 in/out.
// Pre-pass converts K -> f16 (row-major) and V -> f16 TRANSPOSED (V^T [D][S])
// into d_ws (needs 32 MB); the pre-pass stores land in the 192 MB L2, so the
// hot loop reads L2-resident f16 operands directly (no prefetch needed) as
// single 32B-aligned v16h (two adjacent b128). GEMMs on
// v_wmma_f32_16x16x32_f16; row sums via ones-matrix WMMA; tile-wide running
// max; scores in log2 domain (bare v_exp_f32).

#define B_ 2
#define H_ 16
#define S_ 2048
#define D_ 128

typedef __attribute__((ext_vector_type(16))) _Float16 v16h;
typedef __attribute__((ext_vector_type(8)))  _Float16 v8h;
typedef __attribute__((ext_vector_type(8)))  float    v8f;
typedef __attribute__((ext_vector_type(4)))  float    v4f;

constexpr int WAVES  = 8;    // waves per block (LDS only 10 KB)
constexpr int PPITCH = 40;   // P buffer row pitch in halves (80 B, 16B-aligned rows)

#if defined(__has_builtin)
#if __has_builtin(__builtin_amdgcn_exp2f)
#define EXP2(x) __builtin_amdgcn_exp2f(x)   // bare v_exp_f32
#endif
#endif
#ifndef EXP2
#define EXP2(x) exp2f(x)
#endif

__device__ __forceinline__ v16h pack16(v4f a0, v4f a1, v4f b0, v4f b1) {
  v16h r;
#pragma unroll
  for (int e = 0; e < 4; ++e) {
    r[e]      = (_Float16)a0[e];
    r[e + 4]  = (_Float16)a1[e];
    r[e + 8]  = (_Float16)b0[e];
    r[e + 12] = (_Float16)b1[e];
  }
  return r;
}

// ---- Pre-pass 1: K f32 -> f16, same [BH,S,D] layout. b128 in / b128 out. ----
__global__ __launch_bounds__(256) void cvt_k_kernel(const float* __restrict__ in,
                                                    _Float16* __restrict__ out) {
  const size_t i = ((size_t)blockIdx.x * 256 + threadIdx.x) * 8;
  const v4f a = *(const v4f*)(in + i);
  const v4f b = *(const v4f*)(in + i + 4);
  v8h t;
#pragma unroll
  for (int e = 0; e < 4; ++e) { t[e] = (_Float16)a[e]; t[e + 4] = (_Float16)b[e]; }
  *(v8h*)(out + i) = t;
}

// ---- Pre-pass 2: V f32 [BH,S,D] -> f16 V^T [BH,D,S]. Coalesced writes;
//      strided reads are absorbed by L0/L2 (each line reused 128x). ----
__global__ __launch_bounds__(256) void tr_v_kernel(const float* __restrict__ V,
                                                   _Float16* __restrict__ Vt) {
  const int t  = threadIdx.x;
  const int bh = blockIdx.y;
  const int s0 = blockIdx.x * 64;
  const float* src = V + (size_t)bh * S_ * D_;
  _Float16*    dst = Vt + (size_t)bh * D_ * S_;
#pragma unroll 8
  for (int i = 0; i < 32; ++i) {
    const int idx = i * 256 + t;
    const int d   = idx >> 6;
    const int sl  = idx & 63;
    dst[(size_t)d * S_ + s0 + sl] = (_Float16)src[(size_t)(s0 + sl) * D_ + d];
  }
}

__global__ __launch_bounds__(WAVES * 32) void fa_causal_wmma_kernel(
    const float* __restrict__ Q, const _Float16* __restrict__ K16,
    const _Float16* __restrict__ V16t, float* __restrict__ O)
{
  // 1/sqrt(128) * log2(e): scores leave GEMM1 in log2 domain.
  constexpr float SCALE_L2E = 0.08838834764831845f * 1.4426950408889634f;

  const int lane = threadIdx.x & 31;
  const int wave = threadIdx.x >> 5;
  const int half = lane >> 4;
  const int ln   = lane & 15;

  const int bh    = blockIdx.y;
  const int qbase = (blockIdx.x * WAVES + wave) * 16;

  const float*    Qh = Q    + (size_t)bh * S_ * D_;
  const _Float16* Kh = K16  + (size_t)bh * S_ * D_;
  const _Float16* Vt = V16t + (size_t)bh * D_ * S_;   // [D][S]
  float*          Oh = O    + (size_t)bh * S_ * D_;

  // Per-wave LDS: P transpose buffer (C-layout -> A-layout) only.
  __shared__ _Float16 lds_p[WAVES][16 * PPITCH];
  _Float16* pbuf = lds_p[wave];

  // ---- Q tile as 4 f16 A-fragments (16x32), scale*log2e folded in. ----
  // A-layout: lane holds row M = ln; elem e -> K = 16*half + e.
  v16h qa[4];
  {
    const float* qrow = Qh + (size_t)(qbase + ln) * D_;
#pragma unroll
    for (int dc = 0; dc < 4; ++dc) {
      const float* p = qrow + dc * 32 + half * 16;
      v4f a0 = *(const v4f*)(p)      * SCALE_L2E;
      v4f a1 = *(const v4f*)(p + 4)  * SCALE_L2E;
      v4f b0 = *(const v4f*)(p + 8)  * SCALE_L2E;
      v4f b1 = *(const v4f*)(p + 12) * SCALE_L2E;
      qa[dc] = pack16(a0, a1, b0, b1);
    }
  }

  // Ones B-matrix: P x ones gives the row sum broadcast to every lane.
  v16h ones;
#pragma unroll
  for (int e = 0; e < 16; ++e) ones[e] = (_Float16)1.0f;

  // C-layout: VGPR r holds row m = r + 8*half.
  v8f acc[8] = {};
  v8f accl  = {};
  float mrow[8];
#pragma unroll
  for (int r = 0; r < 8; ++r) mrow[r] = -1e30f;

  const int kmax = qbase + 15;

  for (int kv = 0; kv <= kmax; kv += 32) {
    // ---- GEMM1: S = Q * K^T (two 16-key groups), f16 direct from L2. ----
    // B-layout: lane holds key column kv + group + ln; elem e -> K = 16*half + e.
    // 32B-aligned v16h deref -> two adjacent b128 loads, no operand moves.
    v8f c0 = {}, c1 = {};
#pragma unroll
    for (int dc = 0; dc < 4; ++dc) {
      const v16h kb0 = *(const v16h*)(Kh + (size_t)(kv + ln) * D_ + dc * 32 + half * 16);
      c0 = __builtin_amdgcn_wmma_f32_16x16x32_f16(false, qa[dc], false, kb0,
                                                  (short)0, c0, false, false);
      const v16h kb1 = *(const v16h*)(Kh + (size_t)(kv + 16 + ln) * D_ + dc * 32 + half * 16);
      c1 = __builtin_amdgcn_wmma_f32_16x16x32_f16(false, qa[dc], false, kb1,
                                                  (short)0, c1, false, false);
    }

    // ---- Causal mask (diagonal-straddling chunks only). ----
    if (kv + 31 > qbase) {
      const int k0 = kv + ln;
      const int k1 = kv + 16 + ln;
#pragma unroll
      for (int r = 0; r < 8; ++r) {
        const int qr = qbase + r + half * 8;
        c0[r] = (k0 > qr) ? -1e30f : c0[r];
        c1[r] = (k1 > qr) ? -1e30f : c1[r];
      }
    }

    // ---- Tile-wide max (per-row shift-invariance keeps this exact). ----
    float mt = fmaxf(c0[0], c1[0]);
#pragma unroll
    for (int r = 1; r < 8; ++r) mt = fmaxf(mt, fmaxf(c0[r], c1[r]));
    mt = fmaxf(mt, __shfl_xor(mt, 1));
    mt = fmaxf(mt, __shfl_xor(mt, 2));
    mt = fmaxf(mt, __shfl_xor(mt, 4));
    mt = fmaxf(mt, __shfl_xor(mt, 8));
    mt = fmaxf(mt, __shfl_xor(mt, 16));

    // ---- Online rescale + P (f16) into LDS for the C->A transpose. ----
#pragma unroll
    for (int r = 0; r < 8; ++r) {
      const float mnew  = fmaxf(mrow[r], mt);
      const float alpha = EXP2(mrow[r] - mnew);
      const float p0    = EXP2(c0[r]  - mnew);
      const float p1    = EXP2(c1[r]  - mnew);
      mrow[r] = mnew;
      accl[r] *= alpha;
#pragma unroll
      for (int c = 0; c < 8; ++c) acc[c][r] *= alpha;
      const int m = r + half * 8;
      pbuf[m * PPITCH + ln]      = (_Float16)p0;
      pbuf[m * PPITCH + 16 + ln] = (_Float16)p1;
    }

    // Same-wave DS ops are in-order; make the RAW explicit.
    asm volatile("s_wait_dscnt 0" ::: "memory");

    // ---- P back as A-fragment (two aligned ds_load_b128; the two 16B runs
    //      are adjacent, keys 16*half..16*half+15). ----
    const _Float16* prow = pbuf + ln * PPITCH + half * 16;
    const v8h plo = *(const v8h*)(prow);
    const v8h phi = *(const v8h*)(prow + 8);
    const v16h pa = __builtin_shufflevector(plo, phi, 0, 1, 2, 3, 4, 5, 6, 7,
                                            8, 9, 10, 11, 12, 13, 14, 15);

    // Row sums: acc_l = P x ones + acc_l (all 16 columns hold the row sum).
    accl = __builtin_amdgcn_wmma_f32_16x16x32_f16(false, pa, false, ones,
                                                  (short)0, accl, false, false);

    // ---- GEMM2: acc += P * V, B-fragments straight from global V^T f16. ----
    // B-layout: lane holds out-col c*16+ln; elem e -> key kv + 16*half + e.
#pragma unroll
    for (int c = 0; c < 8; ++c) {
      const v16h vb = *(const v16h*)(Vt + (size_t)(c * 16 + ln) * S_ + kv + half * 16);
      acc[c] = __builtin_amdgcn_wmma_f32_16x16x32_f16(false, pa, false, vb,
                                                      (short)0, acc[c], false, false);
    }
  }

  // ---- Normalize and write out (coalesced b32 stores across lanes). ----
  float rl[8];
#pragma unroll
  for (int r = 0; r < 8; ++r) rl[r] = 1.0f / accl[r];
#pragma unroll
  for (int c = 0; c < 8; ++c) {
#pragma unroll
    for (int r = 0; r < 8; ++r) {
      const int m = r + half * 8;
      Oh[(size_t)(qbase + m) * D_ + c * 16 + ln] = acc[c][r] * rl[r];
    }
  }
}

extern "C" void kernel_launch(void* const* d_in, const int* in_sizes, int n_in,
                              void* d_out, int out_size, void* d_ws, size_t ws_size,
                              hipStream_t stream) {
  const float* Q = (const float*)d_in[0];
  const float* K = (const float*)d_in[1];
  const float* V = (const float*)d_in[2];
  // d_in[3] is the tril mask; causality is applied analytically in-kernel.
  float* O = (float*)d_out;

  // Workspace layout (requires 32 MB): [0,16MB) K in f16 row-major,
  // [16MB,32MB) V^T in f16 [BH][D][S]. Re-generated every call (deterministic).
  const size_t nkv = (size_t)B_ * H_ * S_ * D_;
  _Float16* K16  = (_Float16*)d_ws;
  _Float16* V16t = (_Float16*)d_ws + nkv;

  cvt_k_kernel<<<dim3((unsigned)(nkv / (8 * 256))), dim3(256), 0, stream>>>(K, K16);
  tr_v_kernel<<<dim3(S_ / 64, B_ * H_), dim3(256), 0, stream>>>(V, V16t);

  dim3 grid(S_ / (16 * WAVES), B_ * H_);
  fa_causal_wmma_kernel<<<grid, dim3(WAVES * 32), 0, stream>>>(Q, K16, V16t, O);
}
